// MitosisDecoder_3590592659523
// MI455X (gfx1250) — compile-verified, matrix operand order checked by reading, not run on hardware
//
#include <hip/hip_runtime.h>
#include <hip/hip_bf16.h>
#include <math.h>

// ---------------- problem constants ----------------
#define Hdim   1024
#define Bdim   64
#define VP1    32001
#define NPADW  32256          // VP1 padded up to a multiple of 256 (block N tile)
#define NNODES 31
#define DLEV   4

typedef __bf16 bf16_t;
typedef __attribute__((ext_vector_type(16))) __bf16 v16bf;
typedef __attribute__((ext_vector_type(8)))  float  v8f;

union FragU { uint4 u4[2]; v16bf v; };

__device__ inline v16bf load_frag_row(const bf16_t* __restrict__ row, int koff) {
    // lane<16 holds K {koff..koff+7, koff+16..koff+23}; the lane>=16 (+8)
    // shift is folded into koff by the caller.  Two 16-byte loads.
    FragU f;
    f.u4[0] = *reinterpret_cast<const uint4*>(row + koff);
    f.u4[1] = *reinterpret_cast<const uint4*>(row + koff + 16);
    return f.v;
}

// ---------------- bf16 WMMA GEMM ----------------
//   C[M x Nvalid] = A[M x K] * W[Npad x K]^T + bias
// W must be padded (zero rows) up to gridDim.x*256 rows so the K-loop is
// branch-free; Nvalid guards the store epilogue only.
// tree_mode==0: C row m -> Cout + m*ldc
// tree_mode==1: C row m -> dout[ (node_base + (m>>6))*Bdim + (m&63) ]*VP1
__global__ __launch_bounds__(256)
void k_gemm_bf16(const bf16_t* __restrict__ A, const bf16_t* __restrict__ W,
                 const float* __restrict__ bias, float* __restrict__ Cout,
                 int M, int Nvalid, int K, int ldc,
                 int tree_mode, int node_base, float* __restrict__ dout)
{
    const int lane = threadIdx.x & 31;
    const int wave = threadIdx.x >> 5;
    const int wm   = wave & 1;    // 2 M-halves of 32  -> 64 rows / block
    const int wn   = wave >> 1;   // 4 N-strips of 64  -> 256 cols / block
    const int mbase = blockIdx.y * 64  + wm * 32;
    const int nbase = blockIdx.x * 256 + wn * 64;

    const int koff = (lane & 16) ? 8 : 0;
    const bf16_t* A0 = A + (size_t)(mbase +      (lane & 15)) * K;
    const bf16_t* A1 = A + (size_t)(mbase + 16 + (lane & 15)) * K;
    const bf16_t* W0 = W + (size_t)(nbase +      (lane & 15)) * K;
    const bf16_t* W1 = W + (size_t)(nbase + 16 + (lane & 15)) * K;
    const bf16_t* W2 = W + (size_t)(nbase + 32 + (lane & 15)) * K;
    const bf16_t* W3 = W + (size_t)(nbase + 48 + (lane & 15)) * K;

    v8f acc[2][4] = { { {}, {}, {}, {} }, { {}, {}, {}, {} } };

    for (int ks = 0; ks < K; ks += 32) {
        const int kk = ks + koff;
        v16bf a0 = load_frag_row(A0, kk);
        v16bf a1 = load_frag_row(A1, kk);
        v16bf b0 = load_frag_row(W0, kk);
        v16bf b1 = load_frag_row(W1, kk);
        v16bf b2 = load_frag_row(W2, kk);
        v16bf b3 = load_frag_row(W3, kk);
        acc[0][0] = __builtin_amdgcn_wmma_f32_16x16x32_bf16(false, a0, false, b0, (short)0, acc[0][0], false, false);
        acc[1][0] = __builtin_amdgcn_wmma_f32_16x16x32_bf16(false, a1, false, b0, (short)0, acc[1][0], false, false);
        acc[0][1] = __builtin_amdgcn_wmma_f32_16x16x32_bf16(false, a0, false, b1, (short)0, acc[0][1], false, false);
        acc[1][1] = __builtin_amdgcn_wmma_f32_16x16x32_bf16(false, a1, false, b1, (short)0, acc[1][1], false, false);
        acc[0][2] = __builtin_amdgcn_wmma_f32_16x16x32_bf16(false, a0, false, b2, (short)0, acc[0][2], false, false);
        acc[1][2] = __builtin_amdgcn_wmma_f32_16x16x32_bf16(false, a1, false, b2, (short)0, acc[1][2], false, false);
        acc[0][3] = __builtin_amdgcn_wmma_f32_16x16x32_bf16(false, a0, false, b3, (short)0, acc[0][3], false, false);
        acc[1][3] = __builtin_amdgcn_wmma_f32_16x16x32_bf16(false, a1, false, b3, (short)0, acc[1][3], false, false);
    }

    const int hi = (lane >> 4) & 1;
#pragma unroll
    for (int j = 0; j < 4; ++j) {
        const int n0 = nbase + j * 16 + (lane & 15);
        if (n0 >= Nvalid) continue;
        const float bv = bias[n0];
#pragma unroll
        for (int mi = 0; mi < 2; ++mi) {
#pragma unroll
            for (int r = 0; r < 8; ++r) {
                const int m   = mbase + mi * 16 + hi * 8 + r;
                const float v = acc[mi][j][r] + bv;
                if (tree_mode) {
                    const int node = node_base + (m >> 6);
                    dout[((size_t)node * Bdim + (m & 63)) * (size_t)VP1 + n0] = v;
                } else {
                    Cout[(size_t)m * ldc + n0] = v;
                }
            }
        }
    }
}

// ---------------- helpers ----------------
__global__ void k_f32_to_bf16(const float* __restrict__ s, bf16_t* __restrict__ d, size_t n) {
    size_t i = (size_t)blockIdx.x * blockDim.x + threadIdx.x;
    if (i < n) d[i] = (bf16_t)s[i];
}
// convert with zero-padding past n_valid (row padding for Wout)
__global__ void k_f32_to_bf16_pad(const float* __restrict__ s, bf16_t* __restrict__ d,
                                  size_t n_valid, size_t n_total) {
    size_t i = (size_t)blockIdx.x * blockDim.x + threadIdx.x;
    if (i < n_total) d[i] = (i < n_valid) ? (bf16_t)s[i] : (bf16_t)0.0f;
}
__global__ void k_init_h(const float* __restrict__ enc, float* __restrict__ hf,
                         bf16_t* __restrict__ hb, size_t n) {
    size_t i = (size_t)blockIdx.x * blockDim.x + threadIdx.x;
    if (i < n) { float v = enc[i]; hf[i] = v; hb[i] = (bf16_t)v; }
}
__global__ void k_zero_f32(float* __restrict__ p, size_t n) {
    size_t i = (size_t)blockIdx.x * blockDim.x + threadIdx.x;
    if (i < n) p[i] = 0.0f;
}

// argmax over one tree row (first-max-wins, matches jnp.argmax)
__global__ __launch_bounds__(256)
void k_argmax(const float* __restrict__ dout, int prod_base, int* __restrict__ words)
{
    __shared__ float sv[256];
    __shared__ int   si[256];
    const int m    = blockIdx.x;
    const int node = prod_base + (m >> 6);
    const float* row = dout + ((size_t)node * Bdim + (m & 63)) * (size_t)VP1;
    float best = -INFINITY; int bi = VP1;
    for (int c = threadIdx.x; c < VP1; c += 256) {
        float v = row[c];
        if (v > best) { best = v; bi = c; }
    }
    sv[threadIdx.x] = best; si[threadIdx.x] = bi;
    __syncthreads();
    for (int s = 128; s > 0; s >>= 1) {
        if ((int)threadIdx.x < s) {
            float v2 = sv[threadIdx.x + s]; int i2 = si[threadIdx.x + s];
            if (v2 > sv[threadIdx.x] || (v2 == sv[threadIdx.x] && i2 < si[threadIdx.x])) {
                sv[threadIdx.x] = v2; si[threadIdx.x] = i2;
            }
        }
        __syncthreads();
    }
    if (threadIdx.x == 0) words[m] = si[0];
}

__global__ void k_gather_embed(const float* __restrict__ emb, const int* __restrict__ words,
                               bf16_t* __restrict__ x, int Mrows)
{
    int idx = blockIdx.x * blockDim.x + threadIdx.x;
    if (idx >= Mrows * Hdim) return;
    int m = idx >> 10, k = idx & 1023;
    x[idx] = (bf16_t)emb[(size_t)words[m] * Hdim + k];
}

// GRU gates for both children; writes child-ordered hidden state (f32 + bf16)
__global__ void k_gru_gates(const float* __restrict__ gil, const float* __restrict__ ghl,
                            const float* __restrict__ gir, const float* __restrict__ ghr,
                            const float* __restrict__ hin, float* __restrict__ hof,
                            bf16_t* __restrict__ hob, int M)
{
    int idx = blockIdx.x * blockDim.x + threadIdx.x;
    if (idx >= M * Hdim) return;
    const int m = idx >> 10, k = idx & 1023;
    const float h = hin[idx];
    const size_t g = (size_t)m * (3 * Hdim) + k;
    const int i = m >> 6, b = m & 63;
#pragma unroll
    for (int lr = 0; lr < 2; ++lr) {
        const float* gi = lr ? gir : gil;
        const float* gh = lr ? ghr : ghl;
        float r  = 1.0f / (1.0f + __expf(-(gi[g] + gh[g])));
        float z  = 1.0f / (1.0f + __expf(-(gi[g + Hdim] + gh[g + Hdim])));
        float nn = tanhf(gi[g + 2 * Hdim] + r * gh[g + 2 * Hdim]);
        float hp = (1.0f - z) * nn + z * h;
        size_t om = ((size_t)((i << 1) + lr) * Bdim + b) * Hdim + k;
        hof[om] = hp;
        hob[om] = (bf16_t)hp;
    }
}

// in-place row log_softmax on tree slices
__global__ __launch_bounds__(256)
void k_log_softmax(float* __restrict__ dout, int node_base)
{
    __shared__ float sr[256];
    const int m    = blockIdx.x;
    const int node = node_base + (m >> 6);
    float* row = dout + ((size_t)node * Bdim + (m & 63)) * (size_t)VP1;
    float mx = -INFINITY;
    for (int c = threadIdx.x; c < VP1; c += 256) mx = fmaxf(mx, row[c]);
    sr[threadIdx.x] = mx; __syncthreads();
    for (int s = 128; s > 0; s >>= 1) {
        if ((int)threadIdx.x < s) sr[threadIdx.x] = fmaxf(sr[threadIdx.x], sr[threadIdx.x + s]);
        __syncthreads();
    }
    mx = sr[0]; __syncthreads();
    float sum = 0.0f;
    for (int c = threadIdx.x; c < VP1; c += 256) sum += __expf(row[c] - mx);
    sr[threadIdx.x] = sum; __syncthreads();
    for (int s = 128; s > 0; s >>= 1) {
        if ((int)threadIdx.x < s) sr[threadIdx.x] += sr[threadIdx.x + s];
        __syncthreads();
    }
    const float lse = mx + __logf(sr[0]);
    for (int c = threadIdx.x; c < VP1; c += 256) row[c] -= lse;
}

// zero invalid node slices: valid[i] = AND over self+ancestors of (null_rand != 0)
__global__ void k_mask(float* __restrict__ dout, const int* __restrict__ null_rand, size_t total)
{
    size_t idx = (size_t)blockIdx.x * blockDim.x + threadIdx.x;
    if (idx >= total) return;
    int n = (int)(idx / ((size_t)Bdim * VP1));
    bool valid = true;
    for (;;) {
        if (null_rand[n] == 0) { valid = false; break; }
        if (n == 0) break;
        n = (n - 1) >> 1;
    }
    if (!valid) dout[idx] = 0.0f;
}

// ---------------- host launcher ----------------
extern "C" void kernel_launch(void* const* d_in, const int* in_sizes, int n_in,
                              void* d_out, int out_size, void* d_ws, size_t ws_size,
                              hipStream_t stream)
{
    (void)in_sizes; (void)n_in; (void)out_size; (void)ws_size;
    const float* encoding = (const float*)d_in[0];
    const float* emb      = (const float*)d_in[1];
    const float* Wl_ih    = (const float*)d_in[2];
    const float* Wl_hh    = (const float*)d_in[3];
    const float* bl_ih    = (const float*)d_in[4];
    const float* bl_hh    = (const float*)d_in[5];
    const float* Wr_ih    = (const float*)d_in[6];
    const float* Wr_hh    = (const float*)d_in[7];
    const float* br_ih    = (const float*)d_in[8];
    const float* br_hh    = (const float*)d_in[9];
    const float* Wout     = (const float*)d_in[10];
    const float* bout     = (const float*)d_in[11];
    const int*   null_r   = (const int*)d_in[12];
    float* dout = (float*)d_out;

    // -------- workspace carve-up (≈131 MB) --------
    char*  ws  = (char*)d_ws;
    size_t off = 0;
    auto carve = [&](size_t bytes) -> void* {
        off = (off + 255) & ~(size_t)255;
        void* p = ws + off; off += bytes; return p;
    };
    const size_t WGRU = (size_t)3 * Hdim * Hdim;     // 3072*1024 (3072 % 256 == 0)
    bf16_t* wlih_b = (bf16_t*)carve(WGRU * 2);
    bf16_t* wlhh_b = (bf16_t*)carve(WGRU * 2);
    bf16_t* wrih_b = (bf16_t*)carve(WGRU * 2);
    bf16_t* wrhh_b = (bf16_t*)carve(WGRU * 2);
    bf16_t* wout_b = (bf16_t*)carve((size_t)NPADW * Hdim * 2);   // padded rows
    float*  hA_f   = (float*) carve((size_t)1024 * Hdim * 4);
    float*  hB_f   = (float*) carve((size_t)1024 * Hdim * 4);
    bf16_t* hA_b   = (bf16_t*)carve((size_t)1024 * Hdim * 2);
    bf16_t* hB_b   = (bf16_t*)carve((size_t)1024 * Hdim * 2);
    bf16_t* x_b    = (bf16_t*)carve((size_t)512 * Hdim * 2);
    float*  gil    = (float*) carve((size_t)512 * 3 * Hdim * 4);
    float*  ghl    = (float*) carve((size_t)512 * 3 * Hdim * 4);
    float*  gir    = (float*) carve((size_t)512 * 3 * Hdim * 4);
    float*  ghr    = (float*) carve((size_t)512 * 3 * Hdim * 4);
    int*    words  = (int*)   carve((size_t)512 * 4);

    const unsigned T = 256;
    auto blocks = [&](size_t n) -> unsigned { return (unsigned)((n + T - 1) / T); };

    // weights -> bf16 (bf16 Wout = 66 MB, L2-resident afterwards on 192 MB L2)
    k_f32_to_bf16<<<blocks(WGRU), T, 0, stream>>>(Wl_ih, wlih_b, WGRU);
    k_f32_to_bf16<<<blocks(WGRU), T, 0, stream>>>(Wl_hh, wlhh_b, WGRU);
    k_f32_to_bf16<<<blocks(WGRU), T, 0, stream>>>(Wr_ih, wrih_b, WGRU);
    k_f32_to_bf16<<<blocks(WGRU), T, 0, stream>>>(Wr_hh, wrhh_b, WGRU);
    k_f32_to_bf16_pad<<<blocks((size_t)NPADW * Hdim), T, 0, stream>>>(
        Wout, wout_b, (size_t)VP1 * Hdim, (size_t)NPADW * Hdim);

    // node 0 prod = zeros (also what level-0 argmax reads -> word 0)
    k_zero_f32<<<blocks((size_t)Bdim * VP1), T, 0, stream>>>(dout, (size_t)Bdim * VP1);
    // level-0 hidden = encoding
    k_init_h<<<blocks((size_t)Bdim * Hdim), T, 0, stream>>>(encoding, hA_f, hA_b, (size_t)Bdim * Hdim);

    float*  hf = hA_f;  bf16_t* hb = hA_b;
    float*  hf2 = hB_f; bf16_t* hb2 = hB_b;
    int n_l = 1, base = 1, prod_base = 0;

    for (int level = 0; level < DLEV; ++level) {
        const int M = n_l * Bdim;          // 64,128,256,512
        k_argmax<<<M, 256, 0, stream>>>(dout, prod_base, words);
        k_gather_embed<<<blocks((size_t)M * Hdim), T, 0, stream>>>(emb, words, x_b, M);

        dim3 ggru(3 * Hdim / 256, M / 64);  // 12 x (M/64)
        k_gemm_bf16<<<ggru, 256, 0, stream>>>(x_b, wlih_b, bl_ih, gil, M, 3 * Hdim, Hdim, 3 * Hdim, 0, 0, nullptr);
        k_gemm_bf16<<<ggru, 256, 0, stream>>>(hb,  wlhh_b, bl_hh, ghl, M, 3 * Hdim, Hdim, 3 * Hdim, 0, 0, nullptr);
        k_gemm_bf16<<<ggru, 256, 0, stream>>>(x_b, wrih_b, br_ih, gir, M, 3 * Hdim, Hdim, 3 * Hdim, 0, 0, nullptr);
        k_gemm_bf16<<<ggru, 256, 0, stream>>>(hb,  wrhh_b, br_hh, ghr, M, 3 * Hdim, Hdim, 3 * Hdim, 0, 0, nullptr);

        k_gru_gates<<<blocks((size_t)M * Hdim), T, 0, stream>>>(gil, ghl, gir, ghr, hf, hf2, hb2, M);

        const int M2 = 2 * M;              // child rows, in child order
        dim3 gout(NPADW / 256, M2 / 64);   // 126 x (M2/64)
        k_gemm_bf16<<<gout, 256, 0, stream>>>(hb2, wout_b, bout, nullptr, M2, VP1, Hdim, 0, 1, base, dout);
        k_log_softmax<<<M2, 256, 0, stream>>>(dout, base);

        prod_base = base; base += 2 * n_l; n_l <<= 1;
        { float*  t = hf; hf = hf2; hf2 = t; }
        { bf16_t* t = hb; hb = hb2; hb2 = t; }
    }

    const size_t total = (size_t)NNODES * Bdim * VP1;
    k_mask<<<blocks(total), T, 0, stream>>>(dout, null_r, total);
}
